// GEDINeuralProcess_44650480009256
// MI455X (gfx1250) — compile-verified, matrix operand order checked by reading, not run on hardware
//
#include <hip/hip_runtime.h>

#define D   512
#define H   8
#define HD  64
#define NQ  2048
#define NC  4096
#define EPS 1e-5f

typedef __bf16 bf16_t;
typedef __attribute__((ext_vector_type(8)))  __bf16 v8bf;
typedef __attribute__((ext_vector_type(16))) __bf16 v16bf;
typedef __attribute__((ext_vector_type(8)))  float  v8f;
typedef __attribute__((ext_vector_type(4)))  float  v4f;

// Load a 16-element bf16 A/B operand for v_wmma_f32_16x16x32_bf16.
// Per-lane layout (16-bit 16x32 operand): lane L: row=L%16, half=L/16;
// element i -> K = (i<8 ? 0 : 16) + half*8 + (i%8).
// For row-major [row][k] data this is two contiguous 16-byte loads.
static __device__ __forceinline__ v16bf load_ab(const bf16_t* __restrict__ row, int kc, int half) {
  v8bf lo = *(const v8bf*)(row + kc + half * 8);
  v8bf hi = *(const v8bf*)(row + kc + 16 + half * 8);
  return __builtin_shufflevector(lo, hi, 0, 1, 2, 3, 4, 5, 6, 7,
                                 8, 9, 10, 11, 12, 13, 14, 15);
}

static __device__ __forceinline__ v8f wmma_bf16(v16bf a, v16bf b, v8f c) {
  // (neg_a, A, neg_b, B, c_mod, C, reuse_a, reuse_b)
  return __builtin_amdgcn_wmma_f32_16x16x32_bf16(false, a, false, b, (short)0, c, false, false);
}

// ---------------- fp32 -> bf16 convert ----------------
__global__ void __launch_bounds__(256) cvt_bf16_kernel(const float* __restrict__ in,
                                                       bf16_t* __restrict__ out, int n) {
  int i = blockIdx.x * blockDim.x + threadIdx.x;
  if (i < n) out[i] = (bf16_t)in[i];
}

// ---------------- pairwise distances (NQ x NC), stays in L2 ----------------
__global__ void __launch_bounds__(256) dist_kernel(const float* __restrict__ qc,
                                                   const float* __restrict__ cc,
                                                   float* __restrict__ dist) {
  int i = blockIdx.x * blockDim.x + threadIdx.x;  // exact multiple of 256
  int q = i >> 12;        // / NC (4096)
  int k = i & (NC - 1);
  float dx = qc[2 * q]     - cc[2 * k];
  float dy = qc[2 * q + 1] - cc[2 * k + 1];
  dist[i] = sqrtf(dx * dx + dy * dy);
}

// ---------------- C[M,N] = A[M,K] * B[N,K]^T (+bias), bf16 in, bf16 out ----------------
// Block = 256 threads = 8 waves; wave w owns rows [w*16, w*16+16), block owns 64 cols.
// TRANS=true writes Out[col*ldout + row] (used to produce V^T with shape (D, NC)).
template <bool TRANS, bool BIAS>
__global__ void __launch_bounds__(256, 1)
gemm_bf16_kernel(const bf16_t* __restrict__ A, const bf16_t* __restrict__ B,
                 const float* __restrict__ bias, bf16_t* __restrict__ Out,
                 int Kd, int ldout) {
  const int wave = threadIdx.x >> 5;
  const int lane = threadIdx.x & 31;
  const int lm = lane & 15, half = lane >> 4;
  const int mbase = blockIdx.x * 128 + wave * 16;
  const int nbase = blockIdx.y * 64;

  const bf16_t* __restrict__ arow = A + (size_t)(mbase + lm) * Kd;
  v8f acc[4] = {};
  for (int kc = 0; kc < Kd; kc += 32) {
    if (kc + 32 < Kd) __builtin_prefetch(arow + kc + 32, 0, 1);
    v16bf a = load_ab(arow, kc, half);
#pragma unroll
    for (int nt = 0; nt < 4; ++nt) {
      v16bf b = load_ab(B + (size_t)(nbase + nt * 16 + lm) * Kd, kc, half);
      acc[nt] = wmma_bf16(a, b, acc[nt]);
    }
  }
#pragma unroll
  for (int nt = 0; nt < 4; ++nt) {
    int col = nbase + nt * 16 + lm;
    float bv = BIAS ? bias[col] : 0.0f;
#pragma unroll
    for (int r = 0; r < 8; ++r) {
      int row = mbase + r + 8 * half;  // C layout: VGPR r -> M = r + 8*(lane/16)
      float v = acc[nt][r] + bv;
      if (TRANS) Out[(size_t)col * ldout + row] = (bf16_t)v;
      else       Out[(size_t)row * ldout + col] = (bf16_t)v;
    }
  }
}

// ---------------- flash attention, transposed-score formulation ----------------
// One wave per (head, 16-query tile); 4 waves per block. Computes S^T = K Q^T so the
// C-tile has M=context, N=query:
//   * softmax is a per-lane reduction over the lane's own 16 values + ONE shfl_xor(16)
//   * running m/l are per-lane scalars (lane's query column)
//   * the probability tile in C-layout IS the B-operand layout for ctx^T += V^T_tile x P^T
//     (no LDS transpose, no barriers)
//   * epilogue: one contiguous 16B store per 16x16 tile
__global__ void __launch_bounds__(128, 1)
attn_kernel(const bf16_t* __restrict__ Qb, const bf16_t* __restrict__ Kb,
            const bf16_t* __restrict__ Vt, const float* __restrict__ dist,
            const float* __restrict__ log_scale, const float* __restrict__ bias_ph,
            bf16_t* __restrict__ Ctx) {
  const int lane = threadIdx.x & 31;
  const int wave = threadIdx.x >> 5;
  const int lm = lane & 15, half = lane >> 4;
  const int h = blockIdx.y;
  const int qb = (blockIdx.x * 4 + wave) * 16;
  const float sc = 0.125f;  // 1/sqrt(HD=64)
  const float bscale = -__expf(log_scale[0]) * bias_ph[h];

  // Q as the B operand of S^T = K Q^T: lane -> query row qb+lm (loop invariant)
  const bf16_t* __restrict__ qrow = Qb + (size_t)(qb + lm) * D + h * HD;
  v16bf qB0 = load_ab(qrow, 0, half);
  v16bf qB1 = load_ab(qrow, 32, half);

  const float* __restrict__ drow = dist + (size_t)(qb + lm) * NC;  // this lane's query row

  float m = -1e30f, l = 0.0f;  // per-lane (per-query-column) softmax state
  v8f acc[4] = {};             // ctx^T accumulator: M = head-dim, N = query

  for (int kb = 0; kb < NC; kb += 32) {
    // S^T tiles: A = K rows (context x HD), B = Q
    v8f s[2] = {};
#pragma unroll
    for (int t = 0; t < 2; ++t) {
      const bf16_t* __restrict__ krow = Kb + (size_t)(kb + t * 16 + lm) * D + h * HD;
      v16bf ka0 = load_ab(krow, 0, half);
      v16bf ka1 = load_ab(krow, 32, half);
      s[t] = wmma_bf16(ka0, qB0, s[t]);
      s[t] = wmma_bf16(ka1, qB1, s[t]);
    }
    if (kb + 32 < NC) __builtin_prefetch(Kb + (size_t)(kb + 32 + lm) * D + h * HD, 0, 1);

    // scale + distance bias; element (M = k = kb + t*16 + r + 8*half, N = q = qb+lm)
    // k is consecutive in r -> vector loads from the lane's own dist row
#pragma unroll
    for (int t = 0; t < 2; ++t) {
      const float* __restrict__ dp = drow + kb + t * 16 + 8 * half;
      v4f d0 = *(const v4f*)(dp);
      v4f d1 = *(const v4f*)(dp + 4);
#pragma unroll
      for (int r = 0; r < 4; ++r) s[t][r]     = s[t][r]     * sc + bscale * d0[r];
#pragma unroll
      for (int r = 0; r < 4; ++r) s[t][r + 4] = s[t][r + 4] * sc + bscale * d1[r];
    }

    // column max: in-lane over 16 values, then one cross-half exchange
    float mt = s[0][0];
#pragma unroll
    for (int r = 1; r < 8; ++r) mt = fmaxf(mt, s[0][r]);
#pragma unroll
    for (int r = 0; r < 8; ++r) mt = fmaxf(mt, s[1][r]);
    mt = fmaxf(mt, __shfl_xor(mt, 16, 32));

    float mn = fmaxf(m, mt);
    float fs = __expf(m - mn);
    m = mn;

    // probabilities + column sum (in-lane, then one cross-half exchange)
    float ps = 0.0f;
#pragma unroll
    for (int t = 0; t < 2; ++t)
#pragma unroll
      for (int r = 0; r < 8; ++r) {
        float p = __expf(s[t][r] - m);
        s[t][r] = p;
        ps += p;
      }
    ps += __shfl_xor(ps, 16, 32);
    l = l * fs + ps;

    // rescale ctx accumulator by the lane-uniform factor
#pragma unroll
    for (int nt = 0; nt < 4; ++nt)
#pragma unroll
      for (int r = 0; r < 8; ++r) acc[nt][r] *= fs;

    // P^T (C layout, M=context, N=query) is already the B-operand layout:
    // element i -> K = (i<8?0:16) + half*8 + (i%8)  ==  s[i<8?0:1][i%8]
    v16bf pb;
#pragma unroll
    for (int i = 0; i < 8; ++i)  pb[i]     = (bf16_t)s[0][i];
#pragma unroll
    for (int i = 0; i < 8; ++i)  pb[i + 8] = (bf16_t)s[1][i];

    // ctx^T += V^T_tile x P^T   (A = Vt rows = head-dims, K = context chunk)
#pragma unroll
    for (int nt = 0; nt < 4; ++nt) {
      const bf16_t* __restrict__ vrow = Vt + (size_t)(h * HD + nt * 16 + lm) * NC;
      v16bf va = load_ab(vrow, kb, half);
      acc[nt] = wmma_bf16(va, pb, acc[nt]);
    }
  }

  // normalize and store: lane's query row, d = nt*16 + 8*half + r consecutive in r
  const float inv_l = 1.0f / l;
  bf16_t* __restrict__ crow = Ctx + (size_t)(qb + lm) * D + h * HD;
#pragma unroll
  for (int nt = 0; nt < 4; ++nt) {
    v8bf ov;
#pragma unroll
    for (int r = 0; r < 8; ++r) ov[r] = (bf16_t)(acc[nt][r] * inv_l);
    *(v8bf*)(crow + nt * 16 + 8 * half) = ov;
  }
}

// ---------------- out-projection + residual + LayerNorm ----------------
// Block = 256 threads = 8 waves, each block owns 16 query rows x all 512 cols.
__global__ void __launch_bounds__(256, 1)
outproj_ln_kernel(const bf16_t* __restrict__ Ctx, const bf16_t* __restrict__ Wo,
                  const float* __restrict__ bo, const float* __restrict__ resid,
                  const float* __restrict__ ln_g, const float* __restrict__ ln_b,
                  float* __restrict__ out) {
  const int wave = threadIdx.x >> 5;
  const int lane = threadIdx.x & 31;
  const int lm = lane & 15, half = lane >> 4;
  const int mbase = blockIdx.x * 16;
  const int nbase = wave * 64;

  __shared__ float xs[16][D];  // 32 KB

  const bf16_t* __restrict__ arow = Ctx + (size_t)(mbase + lm) * D;
  v8f acc[4] = {};
  for (int kc = 0; kc < D; kc += 32) {
    if (kc + 32 < D) __builtin_prefetch(arow + kc + 32, 0, 1);
    v16bf a = load_ab(arow, kc, half);
#pragma unroll
    for (int nt = 0; nt < 4; ++nt) {
      v16bf b = load_ab(Wo + (size_t)(nbase + nt * 16 + lm) * D, kc, half);
      acc[nt] = wmma_bf16(a, b, acc[nt]);
    }
  }
#pragma unroll
  for (int nt = 0; nt < 4; ++nt) {
    int col = nbase + nt * 16 + lm;
    float bv = bo[col];
#pragma unroll
    for (int r = 0; r < 8; ++r) {
      int rl = r + 8 * half;
      xs[rl][col] = acc[nt][r] + bv + resid[(size_t)(mbase + rl) * D + col];
    }
  }
  __syncthreads();

  // LayerNorm: wave w handles rows 2w and 2w+1; full-wave butterfly reduce
#pragma unroll
  for (int rr = 0; rr < 2; ++rr) {
    int row = wave * 2 + rr;
    float sum = 0.f, sq = 0.f;
#pragma unroll
    for (int j = 0; j < 16; ++j) {
      float v = xs[row][lane * 16 + j];
      sum += v;
      sq  += v * v;
    }
#pragma unroll
    for (int off = 1; off < 32; off <<= 1) {
      sum += __shfl_xor(sum, off, 32);
      sq  += __shfl_xor(sq, off, 32);
    }
    float mean = sum * (1.0f / D);
    float var  = sq * (1.0f / D) - mean * mean;
    float rstd = rsqrtf(var + EPS);
#pragma unroll
    for (int j = 0; j < 16; ++j) {
      int c = lane * 16 + j;
      out[(size_t)(mbase + row) * D + c] = (xs[row][c] - mean) * rstd * ln_g[c] + ln_b[c];
    }
  }
}

extern "C" void kernel_launch(void* const* d_in, const int* in_sizes, int n_in,
                              void* d_out, int out_size, void* d_ws, size_t ws_size,
                              hipStream_t stream) {
  (void)in_sizes; (void)n_in; (void)out_size; (void)ws_size;
  const float* query_repr   = (const float*)d_in[0];
  const float* context_repr = (const float*)d_in[1];
  const float* qcoords      = (const float*)d_in[2];
  const float* ccoords      = (const float*)d_in[3];
  const float* Wq = (const float*)d_in[4];
  const float* bq = (const float*)d_in[5];
  const float* Wk = (const float*)d_in[6];
  const float* bk = (const float*)d_in[7];
  const float* Wv = (const float*)d_in[8];
  const float* bv = (const float*)d_in[9];
  const float* Wo = (const float*)d_in[10];
  const float* bo = (const float*)d_in[11];
  const float* ln_g = (const float*)d_in[12];
  const float* ln_b = (const float*)d_in[13];
  const float* log_scale = (const float*)d_in[14];
  const float* bias_ph   = (const float*)d_in[15];
  float* out = (float*)d_out;

  // workspace layout (all sizes 256B-aligned)
  char* ws = (char*)d_ws;
  size_t off = 0;
  float*  dist = (float*)(ws + off);  off += (size_t)NQ * NC * 4;   // 32 MB
  bf16_t* Xq   = (bf16_t*)(ws + off); off += (size_t)NQ * D * 2;    // 2 MB
  bf16_t* Xc   = (bf16_t*)(ws + off); off += (size_t)NC * D * 2;    // 4 MB
  bf16_t* Wqb  = (bf16_t*)(ws + off); off += (size_t)D * D * 2;
  bf16_t* Wkb  = (bf16_t*)(ws + off); off += (size_t)D * D * 2;
  bf16_t* Wvb  = (bf16_t*)(ws + off); off += (size_t)D * D * 2;
  bf16_t* Wob  = (bf16_t*)(ws + off); off += (size_t)D * D * 2;
  bf16_t* Qb   = (bf16_t*)(ws + off); off += (size_t)NQ * D * 2;
  bf16_t* Kb   = (bf16_t*)(ws + off); off += (size_t)NC * D * 2;
  bf16_t* Vt   = (bf16_t*)(ws + off); off += (size_t)D * NC * 2;    // V transposed (D, NC)
  bf16_t* Ctx  = (bf16_t*)(ws + off); off += (size_t)NQ * D * 2;

  // fp32 -> bf16 converts
  cvt_bf16_kernel<<<(NQ * D + 255) / 256, 256, 0, stream>>>(query_repr, Xq, NQ * D);
  cvt_bf16_kernel<<<(NC * D + 255) / 256, 256, 0, stream>>>(context_repr, Xc, NC * D);
  cvt_bf16_kernel<<<(D * D + 255) / 256, 256, 0, stream>>>(Wq, Wqb, D * D);
  cvt_bf16_kernel<<<(D * D + 255) / 256, 256, 0, stream>>>(Wk, Wkb, D * D);
  cvt_bf16_kernel<<<(D * D + 255) / 256, 256, 0, stream>>>(Wv, Wvb, D * D);
  cvt_bf16_kernel<<<(D * D + 255) / 256, 256, 0, stream>>>(Wo, Wob, D * D);

  // distance matrix (fits in L2)
  dist_kernel<<<(NQ * NC) / 256, 256, 0, stream>>>(qcoords, ccoords, dist);

  // projections: Q = Xq Wq^T + bq ; K = Xc Wk^T + bk ; V^T (stored (D, NC))
  gemm_bf16_kernel<false, true><<<dim3(NQ / 128, D / 64), 256, 0, stream>>>(Xq, Wqb, bq, Qb, D, D);
  gemm_bf16_kernel<false, true><<<dim3(NC / 128, D / 64), 256, 0, stream>>>(Xc, Wkb, bk, Kb, D, D);
  gemm_bf16_kernel<true,  true><<<dim3(NC / 128, D / 64), 256, 0, stream>>>(Xc, Wvb, bv, Vt, D, NC);

  // flash attention: 4 waves/block, one wave per (16-query tile, head)
  attn_kernel<<<dim3(NQ / 64, H), 128, 0, stream>>>(Qb, Kb, Vt, dist, log_scale, bias_ph, Ctx);

  // output projection + residual + LayerNorm
  outproj_ln_kernel<<<NQ / 16, 256, 0, stream>>>(Ctx, Wob, bo, query_repr, ln_g, ln_b, out);
}
